// MoELayer_61014305407659
// MI455X (gfx1250) — compile-verified, matrix operand order checked by reading, not run on hardware
//
#include <hip/hip_runtime.h>
#include <hip/hip_bf16.h>

// Soft-MoE as one f16 WMMA GEMM:
//   Y[b, e*512+i] = coeffs[b,e] * x[b,i]          (B x 8192)
//   Y[b, 8192+e]  = coeffs[b,e]  (e<16), else 0   (bias rider)
//   WT[o, e*512+i] = W[e,i,o]                     (O x 8192)
//   WT[o, 8192+e]  = bias[e,o]   (e<16), else 0
//   out[b,o] = sum_k Y[b,k] * WT[o,k]             (f32 accumulate)

#define B_DIM 1024
#define I_DIM 512
#define O_DIM 512
#define E_DIM 16
#define KTOT  8224   // 16*512 + 32 (bias rider, multiple of 32)
#define NSTEPS (KTOT / 32)

typedef __attribute__((ext_vector_type(16))) _Float16 v16h;
typedef __attribute__((ext_vector_type(8)))  float    v8f;
typedef __attribute__((ext_vector_type(4)))  int      v4i;

union F16Frag {
    v16h v;
    uint4 u[2];
};

#if defined(__gfx1250__) && __has_builtin(__builtin_amdgcn_global_load_async_to_lds_b128) && \
    __has_builtin(__builtin_amdgcn_s_wait_asynccnt)
#define USE_ASYNC_LDS 1
typedef __attribute__((address_space(1))) v4i* gas_v4i_t;  // global <4 x i32>*
typedef __attribute__((address_space(3))) v4i* lds_v4i_t;  // LDS    <4 x i32>*
#else
#define USE_ASYNC_LDS 0
#endif

// ---------------- prep kernels (fp32 -> f16 staging in workspace) -------------

__global__ void prep_y(const float* __restrict__ x, const float* __restrict__ coeffs,
                       _Float16* __restrict__ Y) {
    const int b = blockIdx.x;
    const float* xr = x + (size_t)b * I_DIM;
    const float* cr = coeffs + (size_t)b * E_DIM;
    _Float16* yr = Y + (size_t)b * KTOT;
    for (int k = threadIdx.x; k < KTOT; k += blockDim.x) {
        float v;
        if (k < E_DIM * I_DIM) {
            const int e = k >> 9;
            const int i = k & (I_DIM - 1);
            v = cr[e] * xr[i];
        } else {
            const int e = k - E_DIM * I_DIM;
            v = (e < E_DIM) ? cr[e] : 0.0f;
        }
        yr[k] = (_Float16)v;
    }
}

// W [E,I,O] fp32 -> WT [O, KTOT] f16 with WT[o, e*512+i] = W[e,i,o].
// 32x32 tile transpose through LDS; block = 32x8 threads.
__global__ void prep_wt(const float* __restrict__ W, _Float16* __restrict__ WT) {
    __shared__ float t[32][33];
    const int e  = blockIdx.z;
    const int i0 = blockIdx.y * 32;
    const int o0 = blockIdx.x * 32;
    const int tx = threadIdx.x;
    const int ty = threadIdx.y;
    const float* Wp = W + (size_t)e * I_DIM * O_DIM;
    for (int r = ty; r < 32; r += 8)
        t[r][tx] = Wp[(size_t)(i0 + r) * O_DIM + o0 + tx];
    __syncthreads();
    for (int r = ty; r < 32; r += 8)
        WT[(size_t)(o0 + r) * KTOT + e * I_DIM + i0 + tx] = (_Float16)t[tx][r];
}

__global__ void prep_wt_tail(const float* __restrict__ biases, _Float16* __restrict__ WT) {
    const int o = blockIdx.x * blockDim.x + threadIdx.x;
    if (o >= O_DIM) return;
    _Float16* row = WT + (size_t)o * KTOT + E_DIM * I_DIM;
    for (int e = 0; e < E_DIM; ++e) row[e] = (_Float16)biases[(size_t)e * O_DIM + o];
    for (int e = E_DIM; e < 32; ++e) row[e] = (_Float16)0.0f;
}

// ---------------- main WMMA GEMM ---------------------------------------------
// Block: 256 threads (8 waves), 128x128 output tile. Wave: 64x32 (4x2 WMMAs).
// LDS rows padded to 40 halves (20-dword stride) -> conflict-free b128 reads.
// Double-buffered LDS; tiles staged with GLOBAL_LOAD_ASYNC_TO_LDS_B128
// (ASYNCcnt) when the builtin is available, else through VGPRs.

#define LDS_STRIDE 40

__global__ __launch_bounds__(256) void moe_gemm(const _Float16* __restrict__ Y,
                                                const _Float16* __restrict__ WT,
                                                float* __restrict__ out) {
    __shared__ _Float16 ldsA[2][128 * LDS_STRIDE];
    __shared__ _Float16 ldsB[2][128 * LDS_STRIDE];

    const int tid   = threadIdx.x;
    const int lane  = tid & 31;
    const int w     = tid >> 5;
    const int lm    = lane & 15;
    const int hi    = lane >> 4;
    const int waveM = w >> 2;   // 0..1  -> 64 rows each
    const int waveN = w & 3;    // 0..3  -> 32 cols each
    const int b0    = blockIdx.x * 128;
    const int o0    = blockIdx.y * 128;

    v8f acc[4][2];
#pragma unroll
    for (int i = 0; i < 4; ++i)
#pragma unroll
        for (int j = 0; j < 2; ++j)
            acc[i][j] = (v8f){0.f, 0.f, 0.f, 0.f, 0.f, 0.f, 0.f, 0.f};

    // LDS fill assignments: 512 16-byte chunks per tile, 2 per thread per matrix.
    const int c0   = tid;
    const int c1   = tid + 256;
    const int row0 = c0 >> 2, off0 = (c0 & 3) * 8;
    const int row1 = c1 >> 2, off1 = (c1 & 3) * 8;

    const size_t yBase0 = (size_t)(b0 + row0) * KTOT + off0;
    const size_t yBase1 = (size_t)(b0 + row1) * KTOT + off1;
    const size_t wBase0 = (size_t)(o0 + row0) * KTOT + off0;
    const size_t wBase1 = (size_t)(o0 + row1) * KTOT + off1;

    const int la0 = row0 * LDS_STRIDE + off0;
    const int la1 = row1 * LDS_STRIDE + off1;

#if USE_ASYNC_LDS
    // ---- async double-buffered staging (no VGPR round-trip) ----
#define ASYNC_CP(src, dst)                                           \
    __builtin_amdgcn_global_load_async_to_lds_b128(                  \
        (gas_v4i_t)(void*)(src), (lds_v4i_t)(void*)(dst), 0, 0)

#define ISSUE_TILE(buf, k0)                                          \
    do {                                                             \
        ASYNC_CP(&Y[yBase0 + (k0)],  &ldsA[buf][la0]);               \
        ASYNC_CP(&Y[yBase1 + (k0)],  &ldsA[buf][la1]);               \
        ASYNC_CP(&WT[wBase0 + (k0)], &ldsB[buf][la0]);               \
        ASYNC_CP(&WT[wBase1 + (k0)], &ldsB[buf][la1]);               \
    } while (0)

    ISSUE_TILE(0, 0);
    for (int s = 0; s < NSTEPS; ++s) {
        const int p = s & 1;
        if (s + 1 < NSTEPS) {
            ISSUE_TILE(p ^ 1, (s + 1) * 32);
            __builtin_amdgcn_s_wait_asynccnt(4);  // tile s landed (in-order), tile s+1 in flight
        } else {
            __builtin_amdgcn_s_wait_asynccnt(0);
        }
        __syncthreads();  // publish tile s across all waves

        F16Frag a[4], bfr[2];
#pragma unroll
        for (int i = 0; i < 4; ++i) {
            const int r = waveM * 64 + i * 16 + lm;
            a[i].u[0] = *(const uint4*)&ldsA[p][r * LDS_STRIDE + 8 * hi];
            a[i].u[1] = *(const uint4*)&ldsA[p][r * LDS_STRIDE + 16 + 8 * hi];
        }
#pragma unroll
        for (int j = 0; j < 2; ++j) {
            const int c = waveN * 32 + j * 16 + lm;
            bfr[j].u[0] = *(const uint4*)&ldsB[p][c * LDS_STRIDE + 8 * hi];
            bfr[j].u[1] = *(const uint4*)&ldsB[p][c * LDS_STRIDE + 16 + 8 * hi];
        }

#pragma unroll
        for (int i = 0; i < 4; ++i)
#pragma unroll
            for (int j = 0; j < 2; ++j)
                acc[i][j] = __builtin_amdgcn_wmma_f32_16x16x32_f16(
                    false, a[i].v, false, bfr[j].v,
                    (short)0, acc[i][j], false, false);

        __syncthreads();  // all waves done reading buf p before it is refilled
    }
#undef ISSUE_TILE
#undef ASYNC_CP
#else
    // ---- synchronous staging fallback ----
    for (int s = 0; s < NSTEPS; ++s) {
        const int k0 = s * 32;
        const int p  = s & 1;
        __syncthreads();
        *(uint4*)&ldsA[p][la0] = *(const uint4*)&Y[yBase0 + k0];
        *(uint4*)&ldsA[p][la1] = *(const uint4*)&Y[yBase1 + k0];
        *(uint4*)&ldsB[p][la0] = *(const uint4*)&WT[wBase0 + k0];
        *(uint4*)&ldsB[p][la1] = *(const uint4*)&WT[wBase1 + k0];
        __syncthreads();

        F16Frag a[4], bfr[2];
#pragma unroll
        for (int i = 0; i < 4; ++i) {
            const int r = waveM * 64 + i * 16 + lm;
            a[i].u[0] = *(const uint4*)&ldsA[p][r * LDS_STRIDE + 8 * hi];
            a[i].u[1] = *(const uint4*)&ldsA[p][r * LDS_STRIDE + 16 + 8 * hi];
        }
#pragma unroll
        for (int j = 0; j < 2; ++j) {
            const int c = waveN * 32 + j * 16 + lm;
            bfr[j].u[0] = *(const uint4*)&ldsB[p][c * LDS_STRIDE + 8 * hi];
            bfr[j].u[1] = *(const uint4*)&ldsB[p][c * LDS_STRIDE + 16 + 8 * hi];
        }

#pragma unroll
        for (int i = 0; i < 4; ++i)
#pragma unroll
            for (int j = 0; j < 2; ++j)
                acc[i][j] = __builtin_amdgcn_wmma_f32_16x16x32_f16(
                    false, a[i].v, false, bfr[j].v,
                    (short)0, acc[i][j], false, false);
    }
#endif

    // Epilogue: C/D layout -> VGPR v, lane: M = v + 8*hi, N = lane&15.
#pragma unroll
    for (int i = 0; i < 4; ++i) {
#pragma unroll
        for (int j = 0; j < 2; ++j) {
            const int col   = o0 + waveN * 32 + j * 16 + lm;
            const int mbase = b0 + waveM * 64 + i * 16 + 8 * hi;
#pragma unroll
            for (int v = 0; v < 8; ++v)
                out[(size_t)(mbase + v) * O_DIM + col] = acc[i][j][v];
        }
    }
}

// ---------------- launch ------------------------------------------------------

extern "C" void kernel_launch(void* const* d_in, const int* in_sizes, int n_in,
                              void* d_out, int out_size, void* d_ws, size_t ws_size,
                              hipStream_t stream) {
    const float* x      = (const float*)d_in[0];
    const float* coeffs = (const float*)d_in[1];
    const float* W      = (const float*)d_in[2];
    const float* biases = (const float*)d_in[3];
    float* out          = (float*)d_out;

    _Float16* Y  = (_Float16*)d_ws;                                     // 1024*8224*2 B
    _Float16* WT = (_Float16*)((char*)d_ws + (size_t)B_DIM * KTOT * 2); // 512*8224*2 B

    prep_y<<<B_DIM, 256, 0, stream>>>(x, coeffs, Y);
    prep_wt<<<dim3(O_DIM / 32, I_DIM / 32, E_DIM), dim3(32, 8), 0, stream>>>(W, WT);
    prep_wt_tail<<<(O_DIM + 255) / 256, 256, 0, stream>>>(biases, WT);
    moe_gemm<<<dim3(B_DIM / 128, O_DIM / 128), 256, 0, stream>>>(Y, WT, out);
}
